// ChipHeatmapModel_69200513073729
// MI455X (gfx1250) — matrix operand order repaired and checked by reading, not compiled, not run on hardware
//
#include <hip/hip_runtime.h>

// ---------------------------------------------------------------------------
// CDNA5 / gfx1250 GNN implementation.
// All GEMMs run on V_WMMA_F32_16X16X32_F16 (f16 in, f32 accumulate).
// Activations (h, e) are stored in f16; aggregation buffer is f32 (atomics).
// Weights are pre-packed into the per-lane WMMA B-fragment layout.
// ---------------------------------------------------------------------------

typedef __attribute__((ext_vector_type(16))) _Float16 v16h;
typedef __attribute__((ext_vector_type(8)))  _Float16 v8h;
typedef __attribute__((ext_vector_type(8)))  float    v8f;

#define HDIM 64

// A-fragment (16x32 f16, ISA layout): lane holds row m=lane&15.
// halves 0..7  = K: ktile*32 + hi*8  + (0..7)
// halves 8..15 = K: ktile*32 + 16 + hi*8 + (0..7)
static __device__ __forceinline__ v16h frag_a_f16(const _Float16* rowBase, int ktile, int hi) {
  v8h lo = *(const v8h*)(rowBase + ktile * 32 + hi * 8);
  v8h up = *(const v8h*)(rowBase + ktile * 32 + 16 + hi * 8);
  v16h a;
#pragma unroll
  for (int i = 0; i < 8; ++i) { a[i] = lo[i]; a[i + 8] = up[i]; }
  return a;
}

// Same fragment, sourced from an f32 row (aggregation buffer), converting.
static __device__ __forceinline__ v16h frag_a_f32(const float* rowBase, int ktile, int hi) {
  const float* p0 = rowBase + ktile * 32 + hi * 8;
  const float* p1 = rowBase + ktile * 32 + 16 + hi * 8;
  v16h a;
#pragma unroll
  for (int i = 0; i < 8; ++i) { a[i] = (_Float16)p0[i]; a[i + 8] = (_Float16)p1[i]; }
  return a;
}

// Pre-packed B fragment: [frag][lane][16 halves] contiguous.
static __device__ __forceinline__ v16h frag_b(const _Float16* pack, int frag, int lane) {
  return *(const v16h*)(pack + (size_t)frag * 512 + lane * 16);
}

static __device__ __forceinline__ v8f wmma_f16(v16h a, v16h b, v8f c) {
  return __builtin_amdgcn_wmma_f32_16x16x32_f16(false, a, false, b, (short)0, c, false, false);
}

// ---------------------------------------------------------------------------
// Weight packing: row-major fp32 [K x Ncols] -> f16 B-fragments.
// B 16-bit KxN layout (per ISA): lane l holds column n = l&15; K index =
// (i>>3)*16 + (l>>4)*8 + (i&7) within a 32-deep K tile.
// ---------------------------------------------------------------------------
__global__ void pack_b_kernel(const float* __restrict__ src, _Float16* __restrict__ dst,
                              int K, int Ncols, int ntiles, int totalPerMat, int numMat) {
  int o = blockIdx.x * blockDim.x + threadIdx.x;
  if (o >= totalPerMat * numMat) return;
  int mat  = o / totalPerMat;
  int rem  = o - mat * totalPerMat;
  int i    = rem & 15;
  int lane = (rem >> 4) & 31;
  int fragIdx = rem >> 9;
  int ntile = fragIdx % ntiles;
  int ktile = fragIdx / ntiles;
  int k = ktile * 32 + (i >> 3) * 16 + (lane >> 4) * 8 + (i & 7);
  int n = ntile * 16 + (lane & 15);
  dst[o] = (_Float16)src[(size_t)mat * K * Ncols + (size_t)k * Ncols + n];
}

// ---------------------------------------------------------------------------
// Encoders (tiny K, plain VALU): h = relu(nf @ W2x64 + b), e = relu(ea @ W4x64 + b)
// ---------------------------------------------------------------------------
__global__ void node_enc_kernel(const float* __restrict__ nf, const float* __restrict__ W,
                                const float* __restrict__ b, _Float16* __restrict__ hbuf, int N) {
  int idx = blockIdx.x * blockDim.x + threadIdx.x;   // N*8 threads, 8 outputs each
  if (idx >= N * 8) return;
  int node = idx >> 3;
  int j0 = (idx & 7) * 8;
  float f0 = nf[node * 2 + 0], f1 = nf[node * 2 + 1];
#pragma unroll
  for (int q = 0; q < 8; ++q) {
    int j = j0 + q;
    float v = f0 * W[j] + f1 * W[HDIM + j] + b[j];
    hbuf[(size_t)node * HDIM + j] = (_Float16)(v > 0.f ? v : 0.f);
  }
}

__global__ void edge_enc_kernel(const float* __restrict__ ea, const float* __restrict__ W,
                                const float* __restrict__ b, _Float16* __restrict__ ebuf, int E) {
  int idx = blockIdx.x * blockDim.x + threadIdx.x;   // E*8 threads, 8 outputs each
  if (idx >= E * 8) return;
  int e = idx >> 3;
  int j0 = (idx & 7) * 8;
  float a0 = ea[e * 4 + 0], a1 = ea[e * 4 + 1], a2 = ea[e * 4 + 2], a3 = ea[e * 4 + 3];
#pragma unroll
  for (int q = 0; q < 8; ++q) {
    int j = j0 + q;
    float v = a0 * W[j] + a1 * W[HDIM + j] + a2 * W[2 * HDIM + j] + a3 * W[3 * HDIM + j] + b[j];
    ebuf[(size_t)e * HDIM + j] = (_Float16)(v > 0.f ? v : 0.f);
  }
}

// ---------------------------------------------------------------------------
// Edge MLP: per wave, one 16-edge tile.
// A = [h[src] | h[dst] | e]  (16 x 192, f16), B = packed W_r (192 x 64),
// m = relu(A@B + b); e <- m (in place); aggr[dst] += m (f32 atomics).
// 6 K-tiles x 4 N-tiles = 24 WMMAs per tile.
// ---------------------------------------------------------------------------
__global__ __launch_bounds__(256) void edge_mlp_kernel(
    const _Float16* __restrict__ hbuf, _Float16* __restrict__ ebuf,
    float* __restrict__ aggr, const _Float16* __restrict__ packW,
    const float* __restrict__ bias, const int* __restrict__ srcIdx,
    const int* __restrict__ dstIdx, int numTiles) {
  int wave = (blockIdx.x * blockDim.x + threadIdx.x) >> 5;
  int lane = threadIdx.x & 31;
  if (wave >= numTiles) return;                   // wave-uniform
  int ebase = wave * 16;
  int m = lane & 15, hi = lane >> 4, n = m;
  int edge = ebase + m;
  const _Float16* rowS = hbuf + (size_t)srcIdx[edge] * HDIM;
  const _Float16* rowD = hbuf + (size_t)dstIdx[edge] * HDIM;
  const _Float16* rowE = ebuf + (size_t)edge * HDIM;

  v8f c[4];
#pragma unroll
  for (int j = 0; j < 4; ++j) {
    float bv = bias[j * 16 + n];
#pragma unroll
    for (int g = 0; g < 8; ++g) c[j][g] = bv;
  }

#pragma unroll
  for (int t = 0; t < 6; ++t) {
    const _Float16* rb = (t < 2) ? rowS + t * 32
                        : (t < 4) ? rowD + (t - 2) * 32
                                  : rowE + (t - 4) * 32;
    v16h a = frag_a_f16(rb, 0, hi);
#pragma unroll
    for (int j = 0; j < 4; ++j)
      c[j] = wmma_f16(a, frag_b(packW, t * 4 + j, lane), c[j]);
  }

  // Epilogue: relu -> e (f16, in place) + atomic scatter-add into aggr.
#pragma unroll
  for (int g = 0; g < 8; ++g) {
    int mg = g + hi * 8;
    size_t eRow = (size_t)(ebase + mg) * HDIM;
    size_t aRow = (size_t)dstIdx[ebase + mg] * HDIM;
#pragma unroll
    for (int j = 0; j < 4; ++j) {
      float v = c[j][g];
      v = v > 0.f ? v : 0.f;
      ebuf[eRow + j * 16 + n] = (_Float16)v;
      atomicAdd(&aggr[aRow + j * 16 + n], v);
    }
  }
}

// ---------------------------------------------------------------------------
// Node MLP: per wave, one 16-node tile.
// A = [h | aggr] (16 x 128), h <- relu(A@B + b); aggr tile zeroed for next round.
// ---------------------------------------------------------------------------
__global__ __launch_bounds__(256) void node_mlp_kernel(
    _Float16* __restrict__ hbuf, float* __restrict__ aggr,
    const _Float16* __restrict__ packW, const float* __restrict__ bias, int numTiles) {
  int wave = (blockIdx.x * blockDim.x + threadIdx.x) >> 5;
  int lane = threadIdx.x & 31;
  if (wave >= numTiles) return;
  int nbase = wave * 16;
  int m = lane & 15, hi = lane >> 4, n = m;
  const _Float16* rowH = hbuf + (size_t)(nbase + m) * HDIM;
  const float*    rowA = aggr + (size_t)(nbase + m) * HDIM;

  v8f c[4];
#pragma unroll
  for (int j = 0; j < 4; ++j) {
    float bv = bias[j * 16 + n];
#pragma unroll
    for (int g = 0; g < 8; ++g) c[j][g] = bv;
  }

#pragma unroll
  for (int t = 0; t < 4; ++t) {
    v16h a = (t < 2) ? frag_a_f16(rowH, t, hi) : frag_a_f32(rowA, t - 2, hi);
#pragma unroll
    for (int j = 0; j < 4; ++j)
      c[j] = wmma_f16(a, frag_b(packW, t * 4 + j, lane), c[j]);
  }

#pragma unroll
  for (int g = 0; g < 8; ++g) {
    int mg = g + hi * 8;
    size_t hRow = (size_t)(nbase + mg) * HDIM;
#pragma unroll
    for (int j = 0; j < 4; ++j) {
      float v = c[j][g];
      hbuf[hRow + j * 16 + n] = (_Float16)(v > 0.f ? v : 0.f);
    }
  }
  // Zero this tile's aggregation rows for the next round (loads above already
  // consumed by WMMA, so waits have retired before these stores issue).
  float4* az = (float4*)(aggr + (size_t)nbase * HDIM);
  float4 z; z.x = z.y = z.z = z.w = 0.f;
#pragma unroll
  for (int i = 0; i < 8; ++i) az[lane + i * 32] = z;
}

// ---------------------------------------------------------------------------
// Decoder: emb = h@Wd + bd ; t = relu(emb@W1 + b1) ; logits = t@W2 + b2.
// LDS re-layouts D (C/D layout) back into A layout between chained GEMMs.
// ---------------------------------------------------------------------------
__global__ __launch_bounds__(256) void decode_kernel(
    const _Float16* __restrict__ hbuf,
    const _Float16* __restrict__ pdec, const float* __restrict__ dec_b,
    const _Float16* __restrict__ phm1, const float* __restrict__ hm1_b,
    const _Float16* __restrict__ phm2, const float* __restrict__ hm2_b,
    float* __restrict__ outEmb, float* __restrict__ outLogits, int numTiles) {
  __shared__ __align__(16) _Float16 stage[8][16 * HDIM];
  int wave = (blockIdx.x * blockDim.x + threadIdx.x) >> 5;
  int lane = threadIdx.x & 31;
  if (wave >= numTiles) return;
  _Float16* st = stage[threadIdx.x >> 5];
  int nbase = wave * 16;
  int m = lane & 15, hi = lane >> 4, n = m;
  const _Float16* rowH = hbuf + (size_t)(nbase + m) * HDIM;

  // GEMM 1: emb = h @ dec_W + dec_b
  v8f c[4];
#pragma unroll
  for (int j = 0; j < 4; ++j) {
    float bv = dec_b[j * 16 + n];
#pragma unroll
    for (int g = 0; g < 8; ++g) c[j][g] = bv;
  }
#pragma unroll
  for (int t = 0; t < 2; ++t) {
    v16h a = frag_a_f16(rowH, t, hi);
#pragma unroll
    for (int j = 0; j < 4; ++j)
      c[j] = wmma_f16(a, frag_b(pdec, t * 4 + j, lane), c[j]);
  }
#pragma unroll
  for (int g = 0; g < 8; ++g) {
    int mg = g + hi * 8;
#pragma unroll
    for (int j = 0; j < 4; ++j) {
      float v = c[j][g];
      outEmb[(size_t)(nbase + mg) * HDIM + j * 16 + n] = v;   // fp32 embeddings out
      st[mg * HDIM + j * 16 + n] = (_Float16)v;               // f16 for next GEMM
    }
  }

  // GEMM 2: t = relu(emb @ hm1_W + hm1_b)   (LDS same-wave ops are in order)
  v8f c2[4];
#pragma unroll
  for (int j = 0; j < 4; ++j) {
    float bv = hm1_b[j * 16 + n];
#pragma unroll
    for (int g = 0; g < 8; ++g) c2[j][g] = bv;
  }
#pragma unroll
  for (int t = 0; t < 2; ++t) {
    v16h a = frag_a_f16(st + m * HDIM, t, hi);
#pragma unroll
    for (int j = 0; j < 4; ++j)
      c2[j] = wmma_f16(a, frag_b(phm1, t * 4 + j, lane), c2[j]);
  }
#pragma unroll
  for (int g = 0; g < 8; ++g) {
    int mg = g + hi * 8;
#pragma unroll
    for (int j = 0; j < 4; ++j) {
      float v = c2[j][g];
      st[mg * HDIM + j * 16 + n] = (_Float16)(v > 0.f ? v : 0.f);
    }
  }

  // GEMM 3: logits = t @ hm2_W + hm2_b  (64 N-tiles x 2 K-tiles)
  v16h a0 = frag_a_f16(st + m * HDIM, 0, hi);
  v16h a1 = frag_a_f16(st + m * HDIM, 1, hi);
  for (int j = 0; j < 64; ++j) {
    v8f c3;
    float bv = hm2_b[j * 16 + n];
#pragma unroll
    for (int g = 0; g < 8; ++g) c3[g] = bv;
    c3 = wmma_f16(a0, frag_b(phm2, 0 * 64 + j, lane), c3);
    c3 = wmma_f16(a1, frag_b(phm2, 1 * 64 + j, lane), c3);
#pragma unroll
    for (int g = 0; g < 8; ++g) {
      int mg = g + hi * 8;
      outLogits[(size_t)(nbase + mg) * 1024 + j * 16 + n] = c3[g];
    }
  }
}

// ---------------------------------------------------------------------------
extern "C" void kernel_launch(void* const* d_in, const int* in_sizes, int n_in,
                              void* d_out, int out_size, void* d_ws, size_t ws_size,
                              hipStream_t stream) {
  const float* node_features = (const float*)d_in[0];
  const float* edge_attr     = (const float*)d_in[1];
  const float* node_enc_W    = (const float*)d_in[2];
  const float* node_enc_b    = (const float*)d_in[3];
  const float* edge_enc_W    = (const float*)d_in[4];
  const float* edge_enc_b    = (const float*)d_in[5];
  const float* edge_mlp_W    = (const float*)d_in[6];
  const float* edge_mlp_b    = (const float*)d_in[7];
  const float* node_mlp_W    = (const float*)d_in[8];
  const float* node_mlp_b    = (const float*)d_in[9];
  const float* dec_W         = (const float*)d_in[10];
  const float* dec_b         = (const float*)d_in[11];
  const float* hm1_W         = (const float*)d_in[12];
  const float* hm1_b         = (const float*)d_in[13];
  const float* hm2_W         = (const float*)d_in[14];
  const float* hm2_b         = (const float*)d_in[15];
  const int*   edge_index    = (const int*)d_in[16];

  const int N = in_sizes[0] / 2;
  const int E = in_sizes[1] / 4;
  const int R = in_sizes[7] / HDIM;
  const int G2 = in_sizes[15];

  const int* srcIdx = edge_index;
  const int* dstIdx = edge_index + E;

  // Workspace layout (256B aligned regions).
  char* ws = (char*)d_ws;
  size_t off = 0;
  _Float16* hbuf = (_Float16*)(ws + off); off += (size_t)N * HDIM * 2;        // h (f16)
  off = (off + 255) & ~(size_t)255;
  _Float16* ebuf = (_Float16*)(ws + off); off += (size_t)E * HDIM * 2;        // e (f16)
  off = (off + 255) & ~(size_t)255;
  float*    aggr = (float*)(ws + off);    off += (size_t)N * HDIM * 4;        // aggr (f32)
  off = (off + 255) & ~(size_t)255;
  _Float16* pe   = (_Float16*)(ws + off); off += (size_t)R * 3 * HDIM * HDIM * 2;
  off = (off + 255) & ~(size_t)255;
  _Float16* pn   = (_Float16*)(ws + off); off += (size_t)R * 2 * HDIM * HDIM * 2;
  off = (off + 255) & ~(size_t)255;
  _Float16* pdec = (_Float16*)(ws + off); off += (size_t)HDIM * HDIM * 2;
  off = (off + 255) & ~(size_t)255;
  _Float16* phm1 = (_Float16*)(ws + off); off += (size_t)HDIM * HDIM * 2;
  off = (off + 255) & ~(size_t)255;
  _Float16* phm2 = (_Float16*)(ws + off); off += (size_t)HDIM * G2 * 2;
  (void)ws_size; (void)n_in; (void)out_size;

  hipMemsetAsync(aggr, 0, (size_t)N * HDIM * 4, stream);

  // Pack all weights into WMMA B-fragment layout (f16).
  {
    int tpm = 3 * HDIM * HDIM;  // 192x64
    pack_b_kernel<<<(R * tpm + 255) / 256, 256, 0, stream>>>(edge_mlp_W, pe, 3 * HDIM, HDIM, 4, tpm, R);
  }
  {
    int tpm = 2 * HDIM * HDIM;  // 128x64
    pack_b_kernel<<<(R * tpm + 255) / 256, 256, 0, stream>>>(node_mlp_W, pn, 2 * HDIM, HDIM, 4, tpm, R);
  }
  {
    int tpm = HDIM * HDIM;      // 64x64
    pack_b_kernel<<<(tpm + 255) / 256, 256, 0, stream>>>(dec_W, pdec, HDIM, HDIM, 4, tpm, 1);
    pack_b_kernel<<<(tpm + 255) / 256, 256, 0, stream>>>(hm1_W, phm1, HDIM, HDIM, 4, tpm, 1);
  }
  {
    int tpm = HDIM * G2;        // 64x1024
    pack_b_kernel<<<(tpm + 255) / 256, 256, 0, stream>>>(hm2_W, phm2, HDIM, G2, G2 / 16, tpm, 1);
  }

  // Encoders.
  node_enc_kernel<<<(N * 8 + 255) / 256, 256, 0, stream>>>(node_features, node_enc_W, node_enc_b, hbuf, N);
  edge_enc_kernel<<<(E * 8 + 255) / 256, 256, 0, stream>>>(edge_attr, edge_enc_W, edge_enc_b, ebuf, E);

  // Message-passing rounds.
  int eTiles = E / 16;
  int nTiles = N / 16;
  for (int r = 0; r < R; ++r) {
    edge_mlp_kernel<<<(eTiles + 7) / 8, 256, 0, stream>>>(
        hbuf, ebuf, aggr, pe + (size_t)r * 3 * HDIM * HDIM,
        edge_mlp_b + (size_t)r * HDIM, srcIdx, dstIdx, eTiles);
    node_mlp_kernel<<<(nTiles + 7) / 8, 256, 0, stream>>>(
        hbuf, aggr, pn + (size_t)r * 2 * HDIM * HDIM,
        node_mlp_b + (size_t)r * HDIM, nTiles);
  }

  // Decoder + heatmap head.
  float* outEmb = (float*)d_out;
  float* outLogits = outEmb + (size_t)N * HDIM;
  decode_kernel<<<(nTiles + 7) / 8, 256, 0, stream>>>(
      hbuf, pdec, dec_b, phm1, hm1_b, phm2, hm2_b, outEmb, outLogits, nTiles);
}
